// InvNet_5214090297566
// MI455X (gfx1250) — matrix operand order repaired and checked by reading, not compiled, not run on hardware
//
#include <hip/hip_runtime.h>
#include <math.h>

// ---------------- problem constants (match reference) ----------------
#define BETA_F    0.05f
#define INV_BETA  20.0f
#define KNN       6
#define NSPLIT    10
#define Bdim      256
#define Cdim      32768
#define Fdim      2048
#define SPLITSTEP 205            // ceil(2048/10)

// ---------------- tiling ----------------
#define CTILE   128              // columns of em per workgroup
#define NWGC    (Cdim / CTILE)   // 256 column workgroups
#define NBG     2                // b-groups in grid.y
#define BTILE   (Bdim / NBG)     // 128 rows per workgroup
#define NBT     (BTILE / 16)     // 8 b-tiles per wave
#define STRIDE  129              // padded LDS row stride (bank-conflict free)

typedef __attribute__((ext_vector_type(16))) __bf16 v16bf;
typedef __attribute__((ext_vector_type(8)))  float  v8f;

union BF16x16 { v16bf v; __bf16 e[16]; uint4 q[2]; unsigned int u[8]; };

static __device__ __forceinline__ v8f wmma_bf16(v16bf a, v16bf b, v8f c) {
  // (neg_a, A, neg_b, B, c_mod, C, reuse_a, reuse_b)
  return __builtin_amdgcn_wmma_f32_16x16x32_bf16(false, a, false, b, (short)0, c,
                                                 false, false);
}

// ---------------------------------------------------------------
// A fragment (16x32 bf16) from PRE-CONVERTED bf16 hi/lo arrays.
// lane ln (h=ln>>4, m=ln&15): element e -> k_local = (e<8) ? 8h+e : 16+8h+(e-8)
// Fast path: two 16B vector loads per matrix (k0 multiple of 32 -> 16B aligned).
// ---------------------------------------------------------------
static __device__ __forceinline__ void load_a_frag_fast(
    const __bf16* __restrict__ ahi, const __bf16* __restrict__ alo,
    int row, int k0, v16bf& hi, v16bf& lo) {
  const int lane = threadIdx.x & 31;
  const int h    = lane >> 4;
  const int off  = row * Fdim + k0 + 8 * h;   // element index < 2^19, fits int
  BF16x16 H, L;
  H.q[0] = *reinterpret_cast<const uint4*>(ahi + off);
  H.q[1] = *reinterpret_cast<const uint4*>(ahi + off + 16);
  L.q[0] = *reinterpret_cast<const uint4*>(alo + off);
  L.q[1] = *reinterpret_cast<const uint4*>(alo + off + 16);
  hi = H.v; lo = L.v;
}

// Masked variant for split-boundary chunks: zero elements outside [kmin,kmax).
// BRANCHLESS: full vector loads (always in-bounds), then AND each dword
// (= two k-consecutive bf16) against a range mask (v_cmp/v_cndmask/v_and).
static __device__ __forceinline__ void load_a_frag_masked(
    const __bf16* __restrict__ ahi, const __bf16* __restrict__ alo,
    int row, int k0, int kmin, int kmax, v16bf& hi, v16bf& lo) {
  const int lane = threadIdx.x & 31;
  const int h    = lane >> 4;
  const int off  = row * Fdim + k0 + 8 * h;
  BF16x16 H, L;
  H.q[0] = *reinterpret_cast<const uint4*>(ahi + off);
  H.q[1] = *reinterpret_cast<const uint4*>(ahi + off + 16);
  L.q[0] = *reinterpret_cast<const uint4*>(alo + off);
  L.q[1] = *reinterpret_cast<const uint4*>(alo + off + 16);
#pragma unroll
  for (int d = 0; d < 8; ++d) {
    // dword d holds elements with consecutive k: kg and kg+1
    int kl = (d < 4) ? (8 * h + 2 * d) : (16 + 8 * h + 2 * (d - 4));
    int kg = k0 + kl;
    unsigned int m = ((kg >= kmin) & (kg < kmax)) ? 0x0000FFFFu : 0u;
    m |= ((kg + 1 >= kmin) & (kg + 1 < kmax)) ? 0xFFFF0000u : 0u;
    H.u[d] &= m;
    L.u[d] &= m;
  }
  hi = H.v; lo = L.v;
}

// B fragment: 32x16 bf16, lane ln (h=ln>>4, n=ln&15): element e -> k_local = (e<8) ? 8h+e : 16+8h+(e-8).
// em row c is converted fp32->bf16 hi/lo in-loop (each em element touched twice total).
static __device__ __forceinline__ void load_b_frag(const float* __restrict__ em,
                                                   int c, int k0, v16bf& hi, v16bf& lo) {
  const int lane = threadIdx.x & 31;
  const int h    = lane >> 4;
  const int off  = c * Fdim + k0 + 8 * h;     // element index < 2^26, fits int
  BF16x16 H, L;
#pragma unroll
  for (int e = 0; e < 16; ++e) {
    float v = em[off + e + ((e >> 3) << 3)];
    __bf16 vh = (__bf16)v;
    H.e[e] = vh;
    L.e[e] = (__bf16)(v - (float)vh);
  }
  hi = H.v; lo = L.v;
}

// Spill split partials -> LDS, per-row top-6 -> global, fold into full-sim LDS acc.
static __device__ __forceinline__ void finalize_split(
    v8f (&acc)[NBT], int s, int wgc, int bg, int c0,
    float* __restrict__ splitBuf, float* __restrict__ fullAcc,
    float* __restrict__ cand_sval, int* __restrict__ cand_idx) {
  const int tid  = threadIdx.x;
  const int wave = tid >> 5;
  const int lane = tid & 31;
  const int h    = lane >> 4;
  const int n    = lane & 15;
  const v8f zero = {0, 0, 0, 0, 0, 0, 0, 0};
  // C/D layout: VGPR r, lanes 0-15 -> M=r, lanes 16-31 -> M=r+8; N = lane&15.
#pragma unroll
  for (int bt = 0; bt < NBT; ++bt) {
#pragma unroll
    for (int r = 0; r < 8; ++r) {
      int b = bt * 16 + 8 * h + r;               // 0..BTILE-1
      splitBuf[b * STRIDE + wave * 16 + n] = acc[bt][r];
    }
    acc[bt] = zero;
  }
  __syncthreads();
  if (tid < BTILE) {
    const int b = tid;                            // local row
    float vs[6]; int is[6];
#pragma unroll
    for (int q = 0; q < 6; ++q) { vs[q] = -3.0e38f; is[q] = -1; }
#pragma unroll 1
    for (int j = 0; j < CTILE; ++j) {
      float v = splitBuf[b * STRIDE + j];
      fullAcc[b * STRIDE + j] += v;
      if (v > vs[5]) {
        int p = 5;
        while (p > 0 && v > vs[p - 1]) { vs[p] = vs[p - 1]; is[p] = is[p - 1]; --p; }
        vs[p] = v; is[p] = j;
      }
    }
    const int bglob = bg * BTILE + b;
    size_t base = (((size_t)s * Bdim + bglob) * NWGC + wgc) * 6;
#pragma unroll
    for (int q = 0; q < 6; ++q) {
      cand_sval[base + q] = vs[q];
      cand_idx[base + q]  = c0 + is[q];
    }
  }
  __syncthreads();
}

// --------------------------------------------------------------------
// Phase 0: one-time fp32 -> bf16 hi/lo split of `inputs` (2 MB -> 2x1 MB)
// --------------------------------------------------------------------
extern "C" __global__ void __launch_bounds__(256)
cvt_inputs_kernel(const float* __restrict__ inputs,
                  __bf16* __restrict__ ahi, __bf16* __restrict__ alo) {
  int i = blockIdx.x * 256 + threadIdx.x;      // grid covers Bdim*Fdim
  float v = inputs[i];
  __bf16 h = (__bf16)v;
  ahi[i] = h;
  alo[i] = (__bf16)(v - (float)h);
}

// --------------------------------------------------------------------
// Phase 1: fused split-GEMM (bf16 hi/lo WMMA) + streaming top-6 + lse partials
// grid = (NWGC, NBG) blocks of 256 threads; dynamic LDS = 2 * BTILE * STRIDE floats
// Each wave: one 16-col c-tile, 8 b-tiles -> 8 v8f accumulators (64 VGPRs).
// Loop structure: split-outer / chunk-inner with straight-line head / middle /
// tail sections so the steady-state GEMM loop has no control-flow diamond
// (avoids per-iteration accumulator register rotation).
// --------------------------------------------------------------------
extern "C" __global__ void __launch_bounds__(256)
phase1_kernel(const __bf16* __restrict__ ahi_g, const __bf16* __restrict__ alo_g,
              const float* __restrict__ em, const int* __restrict__ targets,
              float* __restrict__ cand_sval, int* __restrict__ cand_idx,
              float* __restrict__ cand_fval,
              float* __restrict__ lse_max, float* __restrict__ lse_sum,
              float* __restrict__ simt) {
  extern __shared__ float lds[];
  float* splitBuf = lds;                        // BTILE * STRIDE
  float* fullAcc  = lds + BTILE * STRIDE;       // BTILE * STRIDE

  const int wgc  = blockIdx.x;
  const int bg   = blockIdx.y;
  const int c0   = wgc * CTILE;
  const int rb0  = bg * BTILE;                  // global row base
  const int tid  = threadIdx.x;
  const int wave = tid >> 5;
  const int lane = tid & 31;
  const int h    = lane >> 4;
  const int n    = lane & 15;
  const int cw   = c0 + wave * 16 + n;          // em row for this lane's B fragment

  for (int j = tid; j < BTILE * STRIDE; j += 256) fullAcc[j] = 0.0f;
  __syncthreads();

  v8f acc[NBT];
  const v8f zero = {0, 0, 0, 0, 0, 0, 0, 0};
#pragma unroll
  for (int bt = 0; bt < NBT; ++bt) acc[bt] = zero;

#pragma unroll 1
  for (int s = 0; s < NSPLIT; ++s) {
    const int kbeg = SPLITSTEP * s;
    const int nk   = SPLITSTEP * (s + 1);
    const int kend = (nk < Fdim) ? nk : Fdim;
    int k0 = kbeg & ~31;

    // ---- head: partial chunk overlapping previous split (mask k < kbeg) ----
    if (k0 < kbeg) {
      v16bf bhi, blo;
      load_b_frag(em, cw, k0, bhi, blo);
#pragma unroll
      for (int bt = 0; bt < NBT; ++bt) {
        v16bf ahi, alo;
        load_a_frag_masked(ahi_g, alo_g, rb0 + bt * 16 + n, k0, kbeg, Fdim, ahi, alo);
        acc[bt] = wmma_bf16(ahi, bhi, acc[bt]);
        acc[bt] = wmma_bf16(ahi, blo, acc[bt]);
        acc[bt] = wmma_bf16(alo, bhi, acc[bt]);
      }
      k0 += 32;
    }

    // ---- middle: full 32-wide chunks (single straight-line loop body) ----
#pragma unroll 1
    for (; k0 + 32 <= kend; k0 += 32) {
      // Prefetch next K-chunk of this lane's em row (global_prefetch_b8 path);
      // one byte per lane pulls the whole cacheline, rows distinct per lane.
      if (k0 + 32 < Fdim)
        __builtin_prefetch(em + (cw * Fdim + k0 + 32 + 16 * h), 0, 0);
      v16bf bhi, blo;
      load_b_frag(em, cw, k0, bhi, blo);
#pragma unroll
      for (int bt = 0; bt < NBT; ++bt) {
        v16bf ahi, alo;
        load_a_frag_fast(ahi_g, alo_g, rb0 + bt * 16 + n, k0, ahi, alo);
        acc[bt] = wmma_bf16(ahi, bhi, acc[bt]);
        acc[bt] = wmma_bf16(ahi, blo, acc[bt]);
        acc[bt] = wmma_bf16(alo, bhi, acc[bt]);
      }
    }

    // ---- tail: partial chunk overlapping next split (mask k >= kend) ----
    if (k0 < kend) {
      v16bf bhi, blo;
      load_b_frag(em, cw, k0, bhi, blo);
#pragma unroll
      for (int bt = 0; bt < NBT; ++bt) {
        v16bf ahi, alo;
        load_a_frag_masked(ahi_g, alo_g, rb0 + bt * 16 + n, k0, 0, kend, ahi, alo);
        acc[bt] = wmma_bf16(ahi, bhi, acc[bt]);
        acc[bt] = wmma_bf16(ahi, blo, acc[bt]);
        acc[bt] = wmma_bf16(alo, bhi, acc[bt]);
      }
    }

    finalize_split(acc, s, wgc, bg, c0, splitBuf, fullAcc, cand_sval, cand_idx);
  }

  // fullAcc holds raw dot products for this WG's 128 columns x 128 rows.
  if (tid < BTILE) {
    const int b     = tid;
    const int bglob = rb0 + b;
    float mx = -3.0e38f;
#pragma unroll 1
    for (int j = 0; j < CTILE; ++j) mx = fmaxf(mx, fullAcc[b * STRIDE + j]);
    float pm = mx * INV_BETA;
    float ps = 0.0f;
#pragma unroll 1
    for (int j = 0; j < CTILE; ++j)
      ps += __expf(fullAcc[b * STRIDE + j] * INV_BETA - pm);
    lse_max[bglob * NWGC + wgc] = pm;
    lse_sum[bglob * NWGC + wgc] = ps;

#pragma unroll 1
    for (int s = 0; s < NSPLIT; ++s) {
      size_t base = (((size_t)s * Bdim + bglob) * NWGC + wgc) * 6;
#pragma unroll 1
      for (int q = 0; q < 6; ++q) {
        int ci = cand_idx[base + q];            // same-thread store->load, in order
        cand_fval[base + q] = fullAcc[b * STRIDE + (ci - c0)] * INV_BETA;
      }
    }
    int tc = targets[bglob];
    if (tc >= c0 && tc < c0 + CTILE)
      simt[bglob] = fullAcc[b * STRIDE + (tc - c0)] * INV_BETA;
  }
}

// --------------------------------------------------------------------
// Phase 2a: combine per-WG (max, sumexp) partials -> lse[b]
// --------------------------------------------------------------------
extern "C" __global__ void __launch_bounds__(256)
lse_kernel(const float* __restrict__ lse_max, const float* __restrict__ lse_sum,
           float* __restrict__ lse) {
  __shared__ float sm[256];
  __shared__ float ss[256];
  const int b = blockIdx.x, t = threadIdx.x;
  float pm = lse_max[b * NWGC + t];
  float ps = lse_sum[b * NWGC + t];
  sm[t] = pm;
  __syncthreads();
#pragma unroll 1
  for (int off = 128; off > 0; off >>= 1) {
    if (t < off) sm[t] = fmaxf(sm[t], sm[t + off]);
    __syncthreads();
  }
  float gm = sm[0];
  __syncthreads();
  ss[t] = ps * __expf(pm - gm);
  __syncthreads();
#pragma unroll 1
  for (int off = 128; off > 0; off >>= 1) {
    if (t < off) ss[t] += ss[t + off];
    __syncthreads();
  }
  if (t == 0) lse[b] = gm + __logf(ss[0]);
}

// --------------------------------------------------------------------
// Phase 2b: merge candidates -> global top-6 per (s,b) -> loss
// --------------------------------------------------------------------
extern "C" __global__ void zero_out_kernel(float* __restrict__ out) {
  if (blockIdx.x == 0 && threadIdx.x == 0) out[0] = 0.0f;
}

static __device__ __forceinline__ void ins6(float* vs, float* fs, int* is,
                                            float v, float f, int i) {
  if (v <= vs[5]) return;
  int p = 5;
  while (p > 0 && v > vs[p - 1]) {
    vs[p] = vs[p - 1]; fs[p] = fs[p - 1]; is[p] = is[p - 1]; --p;
  }
  vs[p] = v; fs[p] = f; is[p] = i;
}

extern "C" __global__ void __launch_bounds__(64)
loss_kernel(const float* __restrict__ cand_sval, const int* __restrict__ cand_idx,
            const float* __restrict__ cand_fval,
            const float* __restrict__ lse, const float* __restrict__ simt,
            const int* __restrict__ targets, const int* __restrict__ epoch,
            float* __restrict__ out) {
  const int s = blockIdx.x / Bdim;
  const int b = blockIdx.x % Bdim;
  const int ep = epoch[0];
  if (!(KNN > 0 && ep > 4)) {
    if (s == 0 && threadIdx.x == 0)
      atomicAdd(out, -(simt[b] - lse[b]) * (1.0f / (float)Bdim));
    return;
  }
  __shared__ float lvs[64 * 6];
  __shared__ float lfs[64 * 6];
  __shared__ int   lis[64 * 6];
  const int t = threadIdx.x;
  float vs[6], fs[6]; int is[6];
#pragma unroll
  for (int q = 0; q < 6; ++q) { vs[q] = -3.0e38f; fs[q] = 0.0f; is[q] = -1; }
  size_t base = ((size_t)s * Bdim + b) * NWGC * 6;
#pragma unroll 1
  for (int j = t; j < NWGC * 6; j += 64)
    ins6(vs, fs, is, cand_sval[base + j], cand_fval[base + j], cand_idx[base + j]);
#pragma unroll
  for (int q = 0; q < 6; ++q) {
    lvs[t * 6 + q] = vs[q]; lfs[t * 6 + q] = fs[q]; lis[t * 6 + q] = is[q];
  }
  __syncthreads();
  if (t == 0) {
    float gv[6], gf[6]; int gi[6];
#pragma unroll
    for (int q = 0; q < 6; ++q) { gv[q] = -3.0e38f; gf[q] = 0.0f; gi[q] = -1; }
#pragma unroll 1
    for (int j = 0; j < 64 * 6; ++j) ins6(gv, gf, gi, lvs[j], lfs[j], lis[j]);
    const int tgt = targets[b];
    float S = 0.0f; int m = 0;
#pragma unroll
    for (int q = 0; q < 6; ++q) {
      if (gi[q] == tgt) ++m; else S += gf[q];
    }
    float l = lse[b], st = simt[b];
    // -( logp[target] + (1/K) * sum_{c in top6, c != target} logp[c] )
    float contrib = -((st - l) + (1.0f / (float)KNN) * (S - (float)(KNN - m) * l));
    atomicAdd(out, contrib * (1.0f / (float)(NSPLIT * Bdim)));
  }
}

// --------------------------------------------------------------------
// Host launcher
// --------------------------------------------------------------------
extern "C" void kernel_launch(void* const* d_in, const int* in_sizes, int n_in,
                              void* d_out, int out_size, void* d_ws, size_t ws_size,
                              hipStream_t stream) {
  (void)in_sizes; (void)n_in; (void)out_size; (void)ws_size;
  const float* inputs  = (const float*)d_in[0];
  const float* em      = (const float*)d_in[1];
  const int*   targets = (const int*)d_in[2];
  const int*   epoch   = (const int*)d_in[3];
  float* out = (float*)d_out;

  char* ws = (char*)d_ws;
  __bf16* inp_hi = (__bf16*)ws; ws += (size_t)Bdim * Fdim * sizeof(__bf16);
  __bf16* inp_lo = (__bf16*)ws; ws += (size_t)Bdim * Fdim * sizeof(__bf16);
  const size_t candN = (size_t)NSPLIT * Bdim * NWGC * 6;
  float* cand_sval = (float*)ws; ws += candN * sizeof(float);
  float* cand_fval = (float*)ws; ws += candN * sizeof(float);
  int*   cand_idx  = (int*)ws;   ws += candN * sizeof(int);
  float* lse_max   = (float*)ws; ws += (size_t)Bdim * NWGC * sizeof(float);
  float* lse_sum   = (float*)ws; ws += (size_t)Bdim * NWGC * sizeof(float);
  float* simt      = (float*)ws; ws += Bdim * sizeof(float);
  float* lse       = (float*)ws; ws += Bdim * sizeof(float);

  const size_t shmem = (size_t)2 * BTILE * STRIDE * sizeof(float); // ~132 KB of 320 KB LDS
  hipFuncSetAttribute(reinterpret_cast<const void*>(phase1_kernel),
                      hipFuncAttributeMaxDynamicSharedMemorySize, (int)shmem);

  zero_out_kernel<<<1, 64, 0, stream>>>(out);
  cvt_inputs_kernel<<<(Bdim * Fdim) / 256, 256, 0, stream>>>(inputs, inp_hi, inp_lo);
  phase1_kernel<<<dim3(NWGC, NBG), 256, shmem, stream>>>(inp_hi, inp_lo, em, targets,
                                                         cand_sval, cand_idx, cand_fval,
                                                         lse_max, lse_sum, simt);
  lse_kernel<<<Bdim, 256, 0, stream>>>(lse_max, lse_sum, lse);
  loss_kernel<<<NSPLIT * Bdim, 64, 0, stream>>>(cand_sval, cand_idx, cand_fval,
                                                lse, simt, targets, epoch, out);
}
